// Attention_82403242541450
// MI455X (gfx1250) — matrix operand order repaired and checked by reading, not compiled, not run on hardware
//
#include <hip/hip_runtime.h>
#include <hip/hip_bf16.h>

// MI455X / gfx1250 fused causal attention + output projection.
// Round 3: coalesced V-transpose prep, base-2 softmax (v_exp_f32 native),
// K-fragment sharing across q-subtiles, double-buffered async staging in
// BOTH the attention kernel and the projection GEMM.

typedef __attribute__((ext_vector_type(16))) _Float16 v16h;
typedef __attribute__((ext_vector_type(8)))  _Float16 v8h;
typedef __attribute__((ext_vector_type(8)))  float    v8f;

#define SEQ    2048
#define DMODEL 1024
#define NH     16
#define DH     64
#define BATCH  4
#define KPAD   72          // K tile row pitch (halves), 144 B (16B aligned)
#define VPAD   40          // V^T tile row pitch (halves), 80 B (16B aligned)
#define LOG2E  1.44269504088896340736f

union HV16 { v16h v; v8h h2[2]; _Float16 e[16]; };

static __device__ inline unsigned lds_off(const void* p) {
    // generic shared pointer: low 32 bits are the LDS byte offset
    return (unsigned)(unsigned long long)(uintptr_t)p;
}
static __device__ inline void async_b128(unsigned lds, const void* gaddr) {
    asm volatile("global_load_async_to_lds_b128 %0, %1, off"
                 :: "v"(lds), "v"((unsigned long long)(uintptr_t)gaddr)
                 : "memory");
}

// ---------------------------------------------------------------------------
// Prep A: qh = q * (1/sqrt(dh)) * log2(e)  (base-2 softmax domain); kh = k.
// ---------------------------------------------------------------------------
__global__ __launch_bounds__(256) void prep_qk_kernel(
    const float* __restrict__ q, const float* __restrict__ k,
    _Float16* __restrict__ qh, _Float16* __restrict__ kh)
{
    size_t i = (size_t)blockIdx.x * 256 + threadIdx.x;   // [0, B*S*D)
    qh[i] = (_Float16)(q[i] * (0.125f * LOG2E));
    kh[i] = (_Float16)k[i];
}

// ---------------------------------------------------------------------------
// Prep B: vt[b][h][dd][s] = (f16) v[b][s][h*64+dd].
// Block = 256 threads on a 64s x 64dd tile: thread (dd = tid&63, sc = tid>>6).
// Reads coalesced across lanes (dd fastest); writes 32 contiguous bytes.
// ---------------------------------------------------------------------------
__global__ __launch_bounds__(256) void prep_v_kernel(
    const float* __restrict__ v, _Float16* __restrict__ vt)
{
    const int s0 = blockIdx.x * 64;
    const int h  = blockIdx.y;
    const int b  = blockIdx.z;
    const int dd = threadIdx.x & 63;
    const int sc = threadIdx.x >> 6;          // 0..3 -> s chunk of 16

    HV16 pack;
#pragma unroll
    for (int j = 0; j < 16; ++j) {
        int s = s0 + sc * 16 + j;
        pack.e[j] = (_Float16)v[((size_t)(b * SEQ + s)) * DMODEL + h * DH + dd];
    }
    _Float16* drow = vt + ((size_t)((b * NH + h) * DH + dd)) * SEQ + s0 + sc * 16;
    *(v8h*)(drow)     = pack.h2[0];
    *(v8h*)(drow + 8) = pack.h2[1];
}

__global__ __launch_bounds__(256) void prep_w_kernel(
    const float* __restrict__ W, _Float16* __restrict__ Wh)
{
    size_t i = (size_t)blockIdx.x * 256 + threadIdx.x;   // [0, D*D)
    Wh[i] = (_Float16)W[i];
}

// ---------------------------------------------------------------------------
// Attention. Block = 128 threads (4 waves), 128 query rows per block
// (32 per wave as two 16-row sub-tiles). K [32x64] and V^T [64x32] tiles
// double-buffered in LDS via async B128 (5 ops/wave/stage, in-order done ->
// s_wait_asynccnt 0x5 gates the current buffer). S^T = K.Q^T puts the query
// index in the lane -> per-lane scalar softmax state; P re-packs lane-locally
// into the A-operand for O += P.V. Softmax in base-2 (native v_exp_f32).
// ---------------------------------------------------------------------------
__global__ __launch_bounds__(128) void attn_fwd_kernel(
    const _Float16* __restrict__ qh, const _Float16* __restrict__ kh,
    const _Float16* __restrict__ vt, const int* __restrict__ mask,
    float* __restrict__ attn_out)
{
    __shared__ __align__(16) _Float16 Ks[2][32 * KPAD];
    __shared__ __align__(16) _Float16 Vs[2][64 * VPAD];
    __shared__ __align__(16) int      Ms[2][32];

    const int tid  = threadIdx.x;
    const int lane = tid & 31;
    const int wave = tid >> 5;
    const int ln   = lane & 15;
    const int hi   = lane >> 4;

    const int qb0   = blockIdx.x * 128;
    const int h     = blockIdx.y;
    const int b     = blockIdx.z;
    const int qrow0 = qb0 + wave * 32;

    // ---- Q fragments (B-operand): two sub-tiles x two dh-chunks ----
    HV16 Bq[2][2];
#pragma unroll
    for (int t = 0; t < 2; ++t)
#pragma unroll
        for (int c = 0; c < 2; ++c) {
            const _Float16* qp = qh + ((size_t)(b * SEQ + qrow0 + t * 16 + ln)) * DMODEL
                               + h * DH + c * 32 + hi * 8;
            const v8h* p = (const v8h*)qp;
            Bq[t][c].h2[0] = p[0];
            Bq[t][c].h2[1] = p[2];
        }

    v8f Oacc[2][4];
#pragma unroll
    for (int t = 0; t < 2; ++t)
#pragma unroll
        for (int d = 0; d < 4; ++d)
            Oacc[t][d] = (v8f){0.f,0.f,0.f,0.f,0.f,0.f,0.f,0.f};
    float mrun[2] = {-1e30f, -1e30f};
    float lrun[2] = {0.f, 0.f};

    const _Float16* kbase = kh + ((size_t)(b * SEQ)) * DMODEL + h * DH;
    const _Float16* vbase = vt + ((size_t)((b * NH + h) * DH)) * SEQ;
    const int*      mbase = mask + b * SEQ;

    auto issue_stage = [&](int buf, int kv0) {            // 5 async ops / wave
#pragma unroll
        for (int i = 0; i < 2; ++i) {                     // K: 256 x 16B chunks
            int cid = wave * 64 + i * 32 + lane;
            int row = cid >> 3;
            int c8  = (cid & 7) * 8;
            async_b128(lds_off(&Ks[buf][row * KPAD + c8]),
                       kbase + ((size_t)(kv0 + row)) * DMODEL + c8);
        }
#pragma unroll
        for (int i = 0; i < 2; ++i) {                     // V^T: 256 x 16B chunks
            int cid = wave * 64 + i * 32 + lane;
            int d   = cid >> 2;
            int c8  = (cid & 3) * 8;
            async_b128(lds_off(&Vs[buf][d * VPAD + c8]),
                       vbase + (size_t)d * SEQ + kv0 + c8);
        }
        if (lane < 2) {                                   // mask: 8 x 16B chunks
            int cid = wave * 2 + lane;
            async_b128(lds_off(&Ms[buf][cid * 4]), mbase + kv0 + cid * 4);
        }
    };

    const int nkv = (qb0 >> 5) + 4;                       // keys up to qb0+127
    issue_stage(0, 0);

    for (int jb = 0; jb < nkv; ++jb) {
        const int kv0 = jb * 32;
        const int cur = jb & 1;
        if (jb + 1 < nkv) {
            issue_stage(cur ^ 1, kv0 + 32);
            asm volatile("s_wait_asynccnt 0x5" ::: "memory");
        } else {
            asm volatile("s_wait_asynccnt 0x0" ::: "memory");
        }
        __syncthreads();

        if (kv0 <= qrow0 + 31) {                          // any live sub-tile?
            // ---- K A-fragments: shared by both q-sub-tiles ----
            HV16 Ak[2][2];
#pragma unroll
            for (int g = 0; g < 2; ++g) {
                const _Float16* krow = &Ks[cur][(g * 16 + ln) * KPAD];
#pragma unroll
                for (int c = 0; c < 2; ++c) {
                    Ak[g][c].h2[0] = *(const v8h*)(krow + c * 32 + hi * 8);
                    Ak[g][c].h2[1] = *(const v8h*)(krow + c * 32 + 16 + hi * 8);
                }
            }

#pragma unroll
            for (int t = 0; t < 2; ++t) {
                if (kv0 > qrow0 + t * 16 + 15) continue;
                const int qg = qrow0 + t * 16 + ln;

                // ---- S^T tiles ----
                v8f ST[2];
#pragma unroll
                for (int g = 0; g < 2; ++g) {
                    v8f acc = (v8f){0.f,0.f,0.f,0.f,0.f,0.f,0.f,0.f};
#pragma unroll
                    for (int c = 0; c < 2; ++c)
                        acc = __builtin_amdgcn_wmma_f32_16x16x32_f16(
                            false, Ak[g][c].v, false, Bq[t][c].v, (short)0, acc, false, false);
                    ST[g] = acc;
                }

                // ---- mask + online softmax (base-2, per-lane scalar state) --
                float tmax = -__builtin_inff();
#pragma unroll
                for (int g = 0; g < 2; ++g)
#pragma unroll
                    for (int r = 0; r < 8; ++r) {
                        int kloc = g * 16 + r + 8 * hi;
                        int key  = kv0 + kloc;
                        float sv = ST[g][r];
                        bool msk = (key > qg) || (Ms[cur][kloc] != 0);
                        sv = msk ? -__builtin_inff() : sv;
                        ST[g][r] = sv;
                        tmax = fmaxf(tmax, sv);
                    }
                tmax = fmaxf(tmax, __shfl_xor(tmax, 16, 32));
                float mn    = fmaxf(mrun[t], tmax);
                float alpha = __builtin_amdgcn_exp2f(mrun[t] - mn);
                float psum  = 0.f;
                HV16  PA;
#pragma unroll
                for (int r = 0; r < 8; ++r) {
                    float p0 = __builtin_amdgcn_exp2f(ST[0][r] - mn);
                    float p1 = __builtin_amdgcn_exp2f(ST[1][r] - mn);
                    psum += p0 + p1;
                    PA.e[r]     = (_Float16)p0;
                    PA.e[r + 8] = (_Float16)p1;
                }
                psum += __shfl_xor(psum, 16, 32);
                lrun[t] = lrun[t] * alpha + psum;
                mrun[t] = mn;

                float abr[8];
#pragma unroll
                for (int r = 0; r < 8; ++r) abr[r] = __shfl(alpha, r + 24 * hi, 32);
#pragma unroll
                for (int d = 0; d < 4; ++d)
#pragma unroll
                    for (int r = 0; r < 8; ++r) Oacc[t][d][r] *= abr[r];

                // ---- O += P.V ----
#pragma unroll
                for (int d = 0; d < 4; ++d) {
                    const _Float16* vrow = &Vs[cur][(d * 16 + ln) * VPAD];
                    HV16 Bv;
                    Bv.h2[0] = *(const v8h*)(vrow + hi * 8);
                    Bv.h2[1] = *(const v8h*)(vrow + 16 + hi * 8);
                    Oacc[t][d] = __builtin_amdgcn_wmma_f32_16x16x32_f16(
                        false, PA.v, false, Bv.v, (short)0, Oacc[t][d], false, false);
                }
            }
        }
        __syncthreads();
    }

    // ---- normalize + store f32 attention output ----
#pragma unroll
    for (int t = 0; t < 2; ++t) {
        float linv = (lrun[t] > 0.f) ? (1.f / lrun[t]) : 0.f;
        float lbr[8];
#pragma unroll
        for (int r = 0; r < 8; ++r) lbr[r] = __shfl(linv, r + 24 * hi, 32);
#pragma unroll
        for (int d = 0; d < 4; ++d)
#pragma unroll
            for (int r = 0; r < 8; ++r) {
                int qr = qrow0 + t * 16 + r + 8 * hi;
                attn_out[((size_t)(b * SEQ + qr)) * DMODEL + h * DH + d * 16 + ln] =
                    Oacc[t][d][r] * lbr[r];
            }
    }
}

// ---------------------------------------------------------------------------
// Projection: out[8192,1024] = X @ W^T + b. Block = 256 threads (8 waves),
// C tile 128x64, double-buffered k-chunks of 32: W tile via async B128
// (1 op/thread/stage -> s_wait_asynccnt 0x1), X tile f32->f16 cvt-staged.
// ---------------------------------------------------------------------------
__global__ __launch_bounds__(256) void proj_kernel(
    const float* __restrict__ X, const _Float16* __restrict__ Wh,
    const float* __restrict__ bias, float* __restrict__ out)
{
    __shared__ __align__(16) _Float16 As[2][128 * 40];
    __shared__ __align__(16) _Float16 Bs[2][64 * 40];

    const int tid  = threadIdx.x;
    const int lane = tid & 31;
    const int wave = tid >> 5;
    const int ln   = lane & 15;
    const int hi   = lane >> 4;
    const int wm   = wave >> 1;
    const int wn   = wave & 1;
    const int Nbase = blockIdx.x * 64;
    const int Mbase = blockIdx.y * 128;

    auto stage = [&](int buf, int kc) {
        {                                                 // Bs: 256 x 16B async
            int row = tid >> 2;
            int c8  = (tid & 3) * 8;
            async_b128(lds_off(&Bs[buf][row * 40 + c8]),
                       Wh + ((size_t)(Nbase + row)) * DMODEL + kc + c8);
        }
#pragma unroll
        for (int i = 0; i < 4; ++i) {                     // As: f32 -> f16 cvt
            int s   = tid + i * 256;
            int row = s >> 3;
            int c4  = (s & 7) * 4;
            float4 f = *(const float4*)(X + ((size_t)(Mbase + row)) * DMODEL + kc + c4);
            _Float16* d = &As[buf][row * 40 + c4];
            d[0] = (_Float16)f.x; d[1] = (_Float16)f.y;
            d[2] = (_Float16)f.z; d[3] = (_Float16)f.w;
        }
    };

    v8f acc[2][2];
#pragma unroll
    for (int mi = 0; mi < 2; ++mi)
#pragma unroll
        for (int ni = 0; ni < 2; ++ni)
            acc[mi][ni] = (v8f){0.f,0.f,0.f,0.f,0.f,0.f,0.f,0.f};

    stage(0, 0);
    for (int kc = 0; kc < DMODEL; kc += 32) {
        const int cur = (kc >> 5) & 1;
        if (kc + 32 < DMODEL) {
            stage(cur ^ 1, kc + 32);
            asm volatile("s_wait_asynccnt 0x1" ::: "memory");
        } else {
            asm volatile("s_wait_asynccnt 0x0" ::: "memory");
        }
        __syncthreads();

        HV16 Af[2], Bf[2];
#pragma unroll
        for (int mi = 0; mi < 2; ++mi) {
            const _Float16* ar = &As[cur][(wm * 32 + mi * 16 + ln) * 40];
            Af[mi].h2[0] = *(const v8h*)(ar + hi * 8);
            Af[mi].h2[1] = *(const v8h*)(ar + 16 + hi * 8);
        }
#pragma unroll
        for (int ni = 0; ni < 2; ++ni) {
            const _Float16* br = &Bs[cur][(wn * 32 + ni * 16 + ln) * 40];
            Bf[ni].h2[0] = *(const v8h*)(br + hi * 8);
            Bf[ni].h2[1] = *(const v8h*)(br + 16 + hi * 8);
        }
#pragma unroll
        for (int mi = 0; mi < 2; ++mi)
#pragma unroll
            for (int ni = 0; ni < 2; ++ni)
                acc[mi][ni] = __builtin_amdgcn_wmma_f32_16x16x32_f16(
                    false, Af[mi].v, false, Bf[ni].v, (short)0, acc[mi][ni], false, false);
        __syncthreads();
    }

#pragma unroll
    for (int mi = 0; mi < 2; ++mi)
#pragma unroll
        for (int ni = 0; ni < 2; ++ni) {
            int ncol = Nbase + wn * 32 + ni * 16 + ln;
            float bv = bias[ncol];
#pragma unroll
            for (int r = 0; r < 8; ++r) {
                int mrow = Mbase + wm * 32 + mi * 16 + r + 8 * hi;
                out[(size_t)mrow * DMODEL + ncol] = acc[mi][ni][r] + bv;
            }
        }
}

extern "C" void kernel_launch(void* const* d_in, const int* in_sizes, int n_in,
                              void* d_out, int out_size, void* d_ws, size_t ws_size,
                              hipStream_t stream) {
    (void)in_sizes; (void)n_in; (void)out_size; (void)ws_size;
    const float* q    = (const float*)d_in[0];
    const float* k    = (const float*)d_in[1];
    const float* v    = (const float*)d_in[2];
    const float* W    = (const float*)d_in[3];
    const float* bias = (const float*)d_in[4];
    const int*   mask = (const int*)d_in[5];
    float* out = (float*)d_out;

    // workspace layout: [attn f32 32MB][qh 16MB][kh 16MB][vt 16MB][Wh 2MB]
    const size_t NELEM = (size_t)BATCH * SEQ * DMODEL;   // 8M
    float*    attn_ws = (float*)d_ws;
    _Float16* qh = (_Float16*)((char*)d_ws + NELEM * sizeof(float));
    _Float16* kh = qh + NELEM;
    _Float16* vt = kh + NELEM;
    _Float16* Wh = vt + NELEM;

    prep_qk_kernel<<<dim3((unsigned)(NELEM / 256)), 256, 0, stream>>>(q, k, qh, kh);
    prep_v_kernel<<<dim3(SEQ / 64, NH, BATCH), 256, 0, stream>>>(v, vt);
    prep_w_kernel<<<dim3((DMODEL * DMODEL) / 256), 256, 0, stream>>>(W, Wh);

    dim3 g1(SEQ / 128, NH, BATCH);
    attn_fwd_kernel<<<g1, 128, 0, stream>>>(qh, kh, vt, mask, attn_ws);

    dim3 g2(DMODEL / 64, (BATCH * SEQ) / 128);
    proj_kernel<<<g2, 256, 0, stream>>>(attn_ws, Wh, bias, out);
}
